// GCN_80857054314634
// MI455X (gfx1250) — matrix operand order repaired and checked by reading, not compiled, not run on hardware
//
#include <hip/hip_runtime.h>

#define NN 16384
#define FF 64

typedef __attribute__((ext_vector_type(16))) __bf16 v16bf;
typedef __attribute__((ext_vector_type(8)))  float  v8f;
typedef __attribute__((ext_vector_type(4)))  float  v4f;

#define WMMA_BF16(A, B, C) \
    __builtin_amdgcn_wmma_f32_16x16x32_bf16(false, (A), false, (B), (short)0, (C), false, false)

// ---------------- K0: w2t[j][k] = bf16(W2[k][j])  (64x64) ----------------
__global__ void k_w2t(const float* __restrict__ W2, __bf16* __restrict__ w2t) {
    int t = blockIdx.x * blockDim.x + threadIdx.x;      // 4096 threads
    int j = t >> 6, k = t & 63;
    w2t[j * FF + k] = (__bf16)W2[k * FF + j];
}

// ------- K1: sup1T[j][i] = bf16( sum_f x[i][f]*W1[f][j] ); copy w -------
__global__ void k_support1(const float* __restrict__ x, const float* __restrict__ W1,
                           const float* __restrict__ w, __bf16* __restrict__ sup1T,
                           float* __restrict__ out_w) {
    int t = blockIdx.x * blockDim.x + threadIdx.x;      // NN*FF threads
    int i = t >> 6, j = t & 63;
    float s = 0.f;
#pragma unroll
    for (int f = 0; f < FF; ++f) s += x[i * FF + f] * W1[f * FF + j];
    sup1T[(size_t)j * NN + i] = (__bf16)s;
    if (j == 0) out_w[i] = w[i];
}

// One pipeline stage: raw fp32 A row chunk + 4 bf16 B tiles (k-step of 32)
struct Stage {
    v4f  r0, r1, r2, r3;
    v16bf b[4];
};

__device__ __forceinline__ void issue_stage(Stage& s, const float* __restrict__ arow,
                                            const __bf16* const bp[4], int k, int half) {
    const float* ap = arow + k + half * 16;
    __builtin_prefetch(ap + 1024, 0, 3);                // stream-ahead prefetch
    s.r0 = __builtin_nontemporal_load((const v4f*)(ap));
    s.r1 = __builtin_nontemporal_load((const v4f*)(ap + 4));
    s.r2 = __builtin_nontemporal_load((const v4f*)(ap + 8));
    s.r3 = __builtin_nontemporal_load((const v4f*)(ap + 12));
#pragma unroll
    for (int g = 0; g < 4; ++g) s.b[g] = *(const v16bf*)(bp[g] + k);
}

// pack 16 fp32 into a 16x32 bf16 A operand in this code's contiguous k order (same as B)
// (lanes 0-15 -> K 0-15 ; lanes 16-31 -> K 16-31)
__device__ __forceinline__ void consume_stage(const Stage& s, v8f acc[4]) {
    v16bf a;
#pragma unroll
    for (int i2 = 0; i2 < 4; ++i2) {
        a[i2]      = (__bf16)s.r0[i2];
        a[4 + i2]  = (__bf16)s.r1[i2];
        a[8 + i2]  = (__bf16)s.r2[i2];
        a[12 + i2] = (__bf16)s.r3[i2];
    }
    acc[0] = WMMA_BF16(a, s.b[0], acc[0]);
    acc[1] = WMMA_BF16(a, s.b[1], acc[1]);
    acc[2] = WMMA_BF16(a, s.b[2], acc[2]);
    acc[3] = WMMA_BF16(a, s.b[3], acc[3]);
}

// Two-stage ping-pong pipelined 16-row x 64-col strip of adj(fp32 NT) @ supT(bf16).
// Each stage is consumed before its registers are re-issued -> no WAR copies.
// Tail prefetch indices wrap with &(NN-1): redundant re-reads instead of branches.
__device__ __forceinline__ void adj_matmul(const float* __restrict__ arow,
                                           const __bf16* __restrict__ supT,
                                           int l16, int half, v8f acc[4]) {
    const __bf16* bp[4];
#pragma unroll
    for (int g = 0; g < 4; ++g)
        bp[g] = supT + (size_t)(g * 16 + l16) * NN + half * 16;

    Stage s0, s1;
    issue_stage(s0, arow, bp, 0, half);
    issue_stage(s1, arow, bp, 32, half);

    for (int k0 = 0; k0 < NN; k0 += 64) {
        consume_stage(s0, acc);
        issue_stage(s0, arow, bp, (k0 + 64) & (NN - 1), half);
        consume_stage(s1, acc);
        issue_stage(s1, arow, bp, (k0 + 96) & (NN - 1), half);
    }
}

// ---- K2: feat1 = relu(adj @ sup1 + b1); sup2T = (feat1 @ W2)^T in bf16 ----
__global__ void __launch_bounds__(128)
k_conv1(const float* __restrict__ adj, const __bf16* __restrict__ sup1T,
        const float* __restrict__ b1, const __bf16* __restrict__ w2t,
        __bf16* __restrict__ sup2T) {
    __shared__ float feat[4][16][72];                   // 18 KB, per-wave 16x64 tile (padded)
    const int lane = threadIdx.x & 31;
    const int wave = threadIdx.x >> 5;
    const int half = lane >> 4;
    const int l16  = lane & 15;
    const int rowBase = blockIdx.x * 64 + wave * 16;

    v8f acc[4] = {};
    adj_matmul(adj + (size_t)(rowBase + l16) * NN, sup1T, l16, half, acc);

    // bias + relu into LDS (C layout: VGPR r -> M = half*8 + r, N = g*16 + l16)
#pragma unroll
    for (int g = 0; g < 4; ++g) {
        int n = g * 16 + l16;
        float bn = b1[n];
#pragma unroll
        for (int r = 0; r < 8; ++r) {
            float v = acc[g][r] + bn;
            feat[wave][half * 8 + r][n] = v > 0.f ? v : 0.f;
        }
    }
    __syncthreads();

    // second graph-conv support: (16x64 feat) @ (64x64 W2), 8 WMMAs
    v8f acc2[4] = {};
#pragma unroll
    for (int kk = 0; kk < FF; kk += 32) {
        v16bf a2;
#pragma unroll
        for (int i2 = 0; i2 < 8; ++i2) {
            a2[i2]     = (__bf16)feat[wave][l16][kk + half * 16 + i2];
            a2[8 + i2] = (__bf16)feat[wave][l16][kk + half * 16 + 8 + i2];
        }
#pragma unroll
        for (int g = 0; g < 4; ++g) {
            v16bf b = *(const v16bf*)(w2t + (g * 16 + l16) * FF + kk + half * 16);
            acc2[g] = WMMA_BF16(a2, b, acc2[g]);
        }
    }

    // store support2^T bf16
#pragma unroll
    for (int g = 0; g < 4; ++g) {
        int n  = g * 16 + l16;
        int ib = rowBase + half * 8;
#pragma unroll
        for (int r = 0; r < 8; ++r)
            sup2T[(size_t)n * NN + ib + r] = (__bf16)acc2[g][r];
    }
}

// ---- K3: out = (adj @ sup2 + b2) * w ----
__global__ void __launch_bounds__(128)
k_conv2(const float* __restrict__ adj, const __bf16* __restrict__ sup2T,
        const float* __restrict__ b2, const float* __restrict__ wvec,
        float* __restrict__ out) {
    const int lane = threadIdx.x & 31;
    const int wave = threadIdx.x >> 5;
    const int half = lane >> 4;
    const int l16  = lane & 15;
    const int rowBase = blockIdx.x * 64 + wave * 16;

    v8f acc[4] = {};
    adj_matmul(adj + (size_t)(rowBase + l16) * NN, sup2T, l16, half, acc);

#pragma unroll
    for (int g = 0; g < 4; ++g) {
        int n = g * 16 + l16;
        float bn = b2[n];
#pragma unroll
        for (int r = 0; r < 8; ++r) {
            int i = rowBase + half * 8 + r;
            out[(size_t)i * FF + n] = (acc[g][r] + bn) * wvec[i];
        }
    }
}

extern "C" void kernel_launch(void* const* d_in, const int* in_sizes, int n_in,
                              void* d_out, int out_size, void* d_ws, size_t ws_size,
                              hipStream_t stream) {
    (void)in_sizes; (void)n_in; (void)out_size; (void)ws_size;
    const float* x   = (const float*)d_in[0];
    const float* w   = (const float*)d_in[1];
    const float* adj = (const float*)d_in[2];
    const float* W1  = (const float*)d_in[3];
    const float* b1  = (const float*)d_in[4];
    const float* W2  = (const float*)d_in[5];
    const float* b2  = (const float*)d_in[6];
    float* out = (float*)d_out;

    char* ws = (char*)d_ws;                              // ~4.01 MB used
    __bf16* sup1T = (__bf16*)(ws);                       // NN*FF bf16 = 2 MB
    __bf16* sup2T = (__bf16*)(ws + (size_t)NN * FF * 2); // 2 MB
    __bf16* w2t   = (__bf16*)(ws + (size_t)NN * FF * 4); // 8 KB

    k_w2t<<<16, 256, 0, stream>>>(W2, w2t);
    k_support1<<<(NN * FF) / 256, 256, 0, stream>>>(x, W1, w, sup1T, out + (size_t)NN * FF);
    k_conv1<<<NN / 64, 128, 0, stream>>>(adj, sup1T, b1, w2t, sup2T);
    k_conv2<<<NN / 64, 128, 0, stream>>>(adj, sup2T, b2, w, out);
}
